// MoEsparseRouting_28432683500255
// MI455X (gfx1250) — matrix operand order repaired and checked by reading, compiled, not run on hardware
//
#include <hip/hip_runtime.h>
#include <hip/hip_bf16.h>

#define B_  8
#define S_  512
#define D_  768
#define L_  12
#define E_  8
#define H_  12
#define FF_ 3072
#define HD_ 64

typedef __attribute__((ext_vector_type(16))) __bf16 v16bf;
typedef __attribute__((ext_vector_type(8)))  __bf16 v8bf;
typedef __attribute__((ext_vector_type(8)))  float  v8f;

// ---------------------------------------------------------------------------
// Generic batched GEMM:  C = scale * (A[M,K]bf16 @ W[N,K]bf16^T) + bias[n]
// flags: 1 = bf16 output, 2 = transposed store (C[n*ldc+m]), 4 = gelu epilogue
// Batch z decomposed as zo*inner + zi with independent strides.
// Wave tile: 32 x (16*NJ); block = 4 waves (2x2) -> block tile 64 x (32*NJ).
// ---------------------------------------------------------------------------
struct GemmParams {
  const __bf16* A;
  const __bf16* W;
  const float*  bias;
  void*         C;
  int K, lda, ldw, ldc, inner, flags;
  float scale;
  long long sA1, sW1, sC1, sA2, sW2, sC2;
};

__device__ __forceinline__ v16bf load_fragA(const __bf16* p) {
  // A 16-bit 16x32 layout: lane m=lane&15, kb=(lane>>4)*8; k = kb..kb+7 and kb+16..kb+23
  v8bf lo = *(const v8bf*)(p);
  v8bf hi = *(const v8bf*)(p + 16);
  return __builtin_shufflevector(lo, hi, 0,1,2,3,4,5,6,7,8,9,10,11,12,13,14,15);
}

__device__ __forceinline__ v16bf load_fragB(const __bf16* p) {
  // B 16-bit 32x16 layout: lane n=lane&15, kb=(lane>>4)*8; k = kb..kb+7 and kb+16..kb+23
  v8bf lo = *(const v8bf*)(p);
  v8bf hi = *(const v8bf*)(p + 16);
  return __builtin_shufflevector(lo, hi, 0,1,2,3,4,5,6,7,8,9,10,11,12,13,14,15);
}

__device__ __forceinline__ float gelu_tanh(float x) {
  float x3 = x * x * x;
  return 0.5f * x * (1.f + tanhf(0.7978845608028654f * (x + 0.044715f * x3)));
}

template <int NJ>
__global__ __launch_bounds__(128) void gemm_bf16_wmma(GemmParams p) {
  const int z  = blockIdx.z;
  const int zi = (p.inner > 1) ? (z % p.inner) : 0;
  const int zo = (p.inner > 1) ? (z / p.inner) : z;
  const __bf16* A = p.A + (long long)zo * p.sA2 + (long long)zi * p.sA1;
  const __bf16* W = p.W + (long long)zo * p.sW2 + (long long)zi * p.sW1;
  const long long cOff = (long long)zo * p.sC2 + (long long)zi * p.sC1;

  const int wave = threadIdx.x >> 5;
  const int lane = threadIdx.x & 31;
  const int m0 = blockIdx.y * 64 + ((wave >> 1) << 5);
  const int n0 = blockIdx.x * (32 * NJ) + (wave & 1) * (16 * NJ);
  const int fr  = lane & 15;
  const int kbA = (lane >> 4) << 3;   // A: two 8-runs at kbA, kbA+16
  const int kbB = (lane >> 4) << 3;   // B: two 8-runs at kbB, kbB+16

  const __bf16* pa0 = A + (long long)(m0 + fr) * p.lda + kbA;
  const __bf16* pa1 = pa0 + 16LL * p.lda;
  const __bf16* pw0 = W + (long long)(n0 + fr) * p.ldw + kbB;

  v8f acc[2][NJ];
#pragma unroll
  for (int i = 0; i < 2; ++i)
#pragma unroll
    for (int j = 0; j < NJ; ++j) acc[i][j] = (v8f){};

  for (int k0 = 0; k0 < p.K; k0 += 32) {
    v16bf a0 = load_fragA(pa0 + k0);
    v16bf a1 = load_fragA(pa1 + k0);
    v16bf wf[NJ];
#pragma unroll
    for (int j = 0; j < NJ; ++j)
      wf[j] = load_fragB(pw0 + (long long)(16 * j) * p.ldw + k0);
    if (k0 + 32 < p.K) {
      __builtin_prefetch(pa0 + k0 + 32, 0, 3);
      __builtin_prefetch(pw0 + k0 + 32, 0, 3);
    }
#pragma unroll
    for (int j = 0; j < NJ; ++j) {
      acc[0][j] = __builtin_amdgcn_wmma_f32_16x16x32_bf16(false, a0, false, wf[j], (short)0, acc[0][j], false, false);
      acc[1][j] = __builtin_amdgcn_wmma_f32_16x16x32_bf16(false, a1, false, wf[j], (short)0, acc[1][j], false, false);
    }
  }

  // C layout: col n = n0+16j+(lane&15), row m = m0+16i + r + ((lane>>4)*8)
  const int rn     = lane & 15;
  const int rmBase = (lane >> 4) << 3;
  float*  Cf = (float*)p.C + cOff;
  __bf16* Cb = (__bf16*)p.C + cOff;
#pragma unroll
  for (int i = 0; i < 2; ++i)
#pragma unroll
    for (int j = 0; j < NJ; ++j) {
      const int n = n0 + 16 * j + rn;
      const float bias = p.bias ? p.bias[n] : 0.f;
#pragma unroll
      for (int r = 0; r < 8; ++r) {
        const int m = m0 + 16 * i + rmBase + r;
        float v = acc[i][j][r] * p.scale + bias;
        if (p.flags & 4) v = gelu_tanh(v);
        const long long off = (p.flags & 2) ? (long long)n * p.ldc + m
                                            : (long long)m * p.ldc + n;
        if (p.flags & 1) Cb[off] = (__bf16)v;
        else             Cf[off] = v;
      }
    }
}

// ---------------------------------------------------------------------------
// Router: pooled = mean_s(hidden[b]); idx[b] = argmax_e(pooled . router_w[e])
// (softmax is monotone, so argmax of logits == argmax of probs)
// ---------------------------------------------------------------------------
__global__ __launch_bounds__(256) void router_kernel(const float* __restrict__ hs,
                                                     const float* __restrict__ rw,
                                                     int* __restrict__ idx) {
  __shared__ float pooled[D_];
  __shared__ float logits[E_];
  const int b = blockIdx.x;
  for (int d = threadIdx.x; d < D_; d += 256) {
    float s = 0.f;
    const float* p = hs + (long long)b * S_ * D_ + d;
    for (int t = 0; t < S_; ++t) s += p[(long long)t * D_];
    pooled[d] = s * (1.f / (float)S_);
  }
  __syncthreads();
  if (threadIdx.x < E_) {
    float s = 0.f;
    const float* w = rw + threadIdx.x * D_;
    for (int d = 0; d < D_; ++d) s += pooled[d] * w[d];
    logits[threadIdx.x] = s;
  }
  __syncthreads();
  if (threadIdx.x == 0) {
    int best = 0; float bv = logits[0];
    for (int e = 1; e < E_; ++e) if (logits[e] > bv) { bv = logits[e]; best = e; }
    idx[b] = best;
  }
}

// ---------------------------------------------------------------------------
// TT expansion: Weff[b] = bf16( Wbase + 8 * einsum(q0,q1,q2,q3)[idx[b]] )
// out row o = j*32+l2 (j<24,l2<32), col d = n*24+p (n<32,p<24)
// ---------------------------------------------------------------------------
__global__ __launch_bounds__(256) void tt_expand_kernel(
    const float* __restrict__ c0, const float* __restrict__ c1,
    const float* __restrict__ c2, const float* __restrict__ c3,
    const float* __restrict__ Wbase, const int* __restrict__ idx,
    __bf16* __restrict__ Weff, int layer) {
  const int t = blockIdx.x * 256 + threadIdx.x;
  const int b = blockIdx.y;
  const int e = idx[b];
  const int o  = t / D_;
  const int d  = t - o * D_;
  const int j  = o >> 5, l2 = o & 31;
  const int nn = d / 24, pp = d - nn * 24;
  const float* A0 = c0 + ((size_t)e * L_ + layer) * 120 + j * 5;        // [k]
  const float* A1 = c1 + ((size_t)e * L_ + layer) * 800 + l2 * 5;      // [k*160+m]
  const float* A2 = c2 + ((size_t)e * L_ + layer) * 800 + nn * 5;      // [m*160+o2]
  const float* A3 = c3 + ((size_t)e * L_ + layer) * 120 + pp;          // [o2*24]
  float t1[5], t2[5];
#pragma unroll
  for (int m = 0; m < 5; ++m) {
    float s = 0.f;
#pragma unroll
    for (int k = 0; k < 5; ++k) s += A0[k] * A1[k * 160 + m];
    t1[m] = s;
  }
#pragma unroll
  for (int o2 = 0; o2 < 5; ++o2) {
    float s = 0.f;
#pragma unroll
    for (int m = 0; m < 5; ++m) s += t1[m] * A2[m * 160 + o2];
    t2[o2] = s;
  }
  float val = 0.f;
#pragma unroll
  for (int o2 = 0; o2 < 5; ++o2) val += t2[o2] * A3[o2 * 24];
  Weff[(size_t)b * D_ * D_ + t] = (__bf16)(Wbase[t] + 8.0f * val);
}

// ---------------------------------------------------------------------------
// fp32 -> bf16 convert / transpose-convert
// ---------------------------------------------------------------------------
__global__ __launch_bounds__(256) void f32_to_bf16(const float* __restrict__ in,
                                                   __bf16* __restrict__ out, long long n) {
  long long i = (long long)blockIdx.x * 256 + threadIdx.x;
  if (i < n) out[i] = (__bf16)in[i];
}

__global__ __launch_bounds__(256) void transpose_f32_to_bf16(const float* __restrict__ in,
                                                             __bf16* __restrict__ out,
                                                             int R, int C) {
  long long t = (long long)blockIdx.x * 256 + threadIdx.x;
  if (t < (long long)R * C) {
    int r = (int)(t / C), c = (int)(t - (long long)r * C);
    out[(long long)c * R + r] = (__bf16)in[t];
  }
}

// ---------------------------------------------------------------------------
// Row softmax over length 512 (fp32 in -> bf16 out), one row per 128-thr block
// ---------------------------------------------------------------------------
__global__ __launch_bounds__(128) void softmax_rows(const float* __restrict__ s,
                                                    __bf16* __restrict__ p) {
  const long long row = blockIdx.x;
  const float* r = s + row * (long long)S_;
  __bf16*      o = p + row * (long long)S_;
  const int tid = threadIdx.x;
  float v[4], mx = -3.4e38f;
#pragma unroll
  for (int i = 0; i < 4; ++i) { v[i] = r[tid + 128 * i]; mx = fmaxf(mx, v[i]); }
#pragma unroll
  for (int off = 16; off; off >>= 1) mx = fmaxf(mx, __shfl_xor(mx, off, 32));
  __shared__ float red[8];
  const int wv = tid >> 5, ln = tid & 31;
  if (!ln) red[wv] = mx;
  __syncthreads();
  mx = fmaxf(fmaxf(red[0], red[1]), fmaxf(red[2], red[3]));
  float sum = 0.f;
#pragma unroll
  for (int i = 0; i < 4; ++i) { v[i] = __expf(v[i] - mx); sum += v[i]; }
#pragma unroll
  for (int off = 16; off; off >>= 1) sum += __shfl_xor(sum, off, 32);
  if (!ln) red[4 + wv] = sum;
  __syncthreads();
  const float inv = 1.f / (red[4] + red[5] + red[6] + red[7]);
#pragma unroll
  for (int i = 0; i < 4; ++i) o[tid + 128 * i] = (__bf16)(v[i] * inv);
}

// ---------------------------------------------------------------------------
// Fused residual + LayerNorm over D=768; writes fp32 x (in place) + bf16 copy
// ---------------------------------------------------------------------------
__global__ __launch_bounds__(256) void ln_residual(float* __restrict__ x,
                                                   const float* __restrict__ y,
                                                   const float* __restrict__ g,
                                                   const float* __restrict__ b,
                                                   __bf16* __restrict__ x16) {
  const long long row = blockIdx.x;
  float*       xr = x   + row * (long long)D_;
  const float* yr = y   + row * (long long)D_;
  __bf16*      xo = x16 + row * (long long)D_;
  const int tid = threadIdx.x;
  float v[3], s = 0.f, ss = 0.f;
#pragma unroll
  for (int i = 0; i < 3; ++i) {
    const int c = tid + 256 * i;
    float t = xr[c] + yr[c];
    v[i] = t; s += t; ss += t * t;
  }
#pragma unroll
  for (int off = 16; off; off >>= 1) { s += __shfl_xor(s, off, 32); ss += __shfl_xor(ss, off, 32); }
  __shared__ float red[16];
  const int wv = tid >> 5, ln = tid & 31;
  if (!ln) { red[wv] = s; red[8 + wv] = ss; }
  __syncthreads();
  float ts = 0.f, tss = 0.f;
#pragma unroll
  for (int i = 0; i < 8; ++i) { ts += red[i]; tss += red[8 + i]; }
  const float mean = ts * (1.f / (float)D_);
  const float var  = tss * (1.f / (float)D_) - mean * mean;
  const float rstd = rsqrtf(var + 1e-12f);
#pragma unroll
  for (int i = 0; i < 3; ++i) {
    const int c = tid + 256 * i;
    float o = (v[i] - mean) * rstd * g[c] + b[c];
    xr[c] = o;
    xo[c] = (__bf16)o;
  }
}

// ---------------------------------------------------------------------------
// Host side
// ---------------------------------------------------------------------------
static void launch_gemm(hipStream_t s, const __bf16* A, const __bf16* W, const float* bias,
                        void* C, int M, int N, int K, int lda, int ldw, int ldc,
                        int inner, long long sA1, long long sW1, long long sC1,
                        long long sA2, long long sW2, long long sC2,
                        int nbatch, float scale, int flags, int nj) {
  GemmParams p;
  p.A = A; p.W = W; p.bias = bias; p.C = C;
  p.K = K; p.lda = lda; p.ldw = ldw; p.ldc = ldc;
  p.inner = inner; p.flags = flags; p.scale = scale;
  p.sA1 = sA1; p.sW1 = sW1; p.sC1 = sC1; p.sA2 = sA2; p.sW2 = sW2; p.sC2 = sC2;
  dim3 grid(N / (32 * nj), M / 64, nbatch);
  if (nj == 2) gemm_bf16_wmma<2><<<grid, 128, 0, s>>>(p);
  else         gemm_bf16_wmma<4><<<grid, 128, 0, s>>>(p);
}

static inline char* ws_alloc(char*& w, size_t bytes) {
  char* p = w;
  w += (bytes + 255) & ~(size_t)255;
  return p;
}

extern "C" void kernel_launch(void* const* d_in, const int* in_sizes, int n_in,
                              void* d_out, int out_size, void* d_ws, size_t ws_size,
                              hipStream_t stream) {
  (void)in_sizes; (void)n_in; (void)out_size; (void)ws_size;
  const float* hs = (const float*)d_in[0];
  const float* rw = (const float*)d_in[1];
  const float* q0 = (const float*)d_in[2];
  const float* q1 = (const float*)d_in[3];
  const float* q2 = (const float*)d_in[4];
  const float* q3 = (const float*)d_in[5];
  const float* v0 = (const float*)d_in[6];
  const float* v1 = (const float*)d_in[7];
  const float* v2 = (const float*)d_in[8];
  const float* v3 = (const float*)d_in[9];
  const float* Wq = (const float*)d_in[10];
  const float* Wk = (const float*)d_in[11];
  const float* Wv = (const float*)d_in[12];
  const float* Wo = (const float*)d_in[13];
  const float* bq = (const float*)d_in[14];
  const float* bk = (const float*)d_in[15];
  const float* bv = (const float*)d_in[16];
  const float* bo = (const float*)d_in[17];
  const float* W1 = (const float*)d_in[18];
  const float* b1 = (const float*)d_in[19];
  const float* W2 = (const float*)d_in[20];
  const float* b2 = (const float*)d_in[21];
  const float* g1 = (const float*)d_in[22];
  const float* be1 = (const float*)d_in[23];
  const float* g2 = (const float*)d_in[24];
  const float* be2 = (const float*)d_in[25];

  char* w = (char*)d_ws;
  int*    idx    = (int*)   ws_alloc(w, B_ * sizeof(int));
  float*  x      = (float*) ws_alloc(w, (size_t)B_ * S_ * D_ * 4);
  float*  y      = (float*) ws_alloc(w, (size_t)B_ * S_ * D_ * 4);
  __bf16* xb     = (__bf16*)ws_alloc(w, (size_t)B_ * S_ * D_ * 2);
  __bf16* qb     = (__bf16*)ws_alloc(w, (size_t)B_ * S_ * D_ * 2);
  __bf16* kb     = (__bf16*)ws_alloc(w, (size_t)B_ * S_ * D_ * 2);
  __bf16* vT     = (__bf16*)ws_alloc(w, (size_t)B_ * S_ * D_ * 2);  // [b, d, s]
  __bf16* ctxb   = (__bf16*)ws_alloc(w, (size_t)B_ * S_ * D_ * 2);
  float*  scores = (float*) ws_alloc(w, (size_t)B_ * H_ * S_ * S_ * 4);
  __bf16* attnb  = (__bf16*)ws_alloc(w, (size_t)B_ * H_ * S_ * S_ * 2);
  __bf16* hb     = (__bf16*)ws_alloc(w, (size_t)B_ * S_ * FF_ * 2);
  __bf16* Wqe    = (__bf16*)ws_alloc(w, (size_t)B_ * D_ * D_ * 2);
  __bf16* Wve    = (__bf16*)ws_alloc(w, (size_t)B_ * D_ * D_ * 2);
  __bf16* Wk16   = (__bf16*)ws_alloc(w, (size_t)D_ * D_ * 2);
  __bf16* Wo16   = (__bf16*)ws_alloc(w, (size_t)D_ * D_ * 2);
  __bf16* W1T    = (__bf16*)ws_alloc(w, (size_t)D_ * FF_ * 2);      // [FF, D]
  __bf16* W2T    = (__bf16*)ws_alloc(w, (size_t)FF_ * D_ * 2);      // [D, FF]

  const long long SD = (long long)S_ * D_;
  const long long DD = (long long)D_ * D_;
  const long long SS = (long long)S_ * S_;

  router_kernel<<<B_, 256, 0, stream>>>(hs, rw, idx);
  hipMemcpyAsync(x, hs, (size_t)B_ * S_ * D_ * 4, hipMemcpyDeviceToDevice, stream);
  f32_to_bf16<<<(B_ * S_ * D_) / 256, 256, 0, stream>>>(x, xb, (long long)B_ * S_ * D_);

  for (int l = 0; l < L_; ++l) {
    tt_expand_kernel<<<dim3(D_ * D_ / 256, B_), 256, 0, stream>>>(
        q0, q1, q2, q3, Wq + (size_t)l * DD, idx, Wqe, l);
    tt_expand_kernel<<<dim3(D_ * D_ / 256, B_), 256, 0, stream>>>(
        v0, v1, v2, v3, Wv + (size_t)l * DD, idx, Wve, l);
    f32_to_bf16<<<(D_ * D_) / 256, 256, 0, stream>>>(Wk + (size_t)l * DD, Wk16, DD);
    f32_to_bf16<<<(D_ * D_) / 256, 256, 0, stream>>>(Wo + (size_t)l * DD, Wo16, DD);
    transpose_f32_to_bf16<<<(D_ * FF_) / 256, 256, 0, stream>>>(W1 + (size_t)l * D_ * FF_, W1T, D_, FF_);
    transpose_f32_to_bf16<<<(D_ * FF_) / 256, 256, 0, stream>>>(W2 + (size_t)l * FF_ * D_, W2T, FF_, D_);

    // q = x @ Wq_eff[b]^T + bq   -> bf16 [b,s,d]
    launch_gemm(stream, xb, Wqe, bq + l * D_, qb, S_, D_, D_, D_, D_, D_,
                1, 0, 0, 0, SD, DD, SD, B_, 1.f, 1, 4);
    // k = x @ Wk^T + bk          -> bf16 [b,s,d]
    launch_gemm(stream, xb, Wk16, bk + l * D_, kb, S_, D_, D_, D_, D_, D_,
                1, 0, 0, 0, SD, 0, SD, B_, 1.f, 1, 4);
    // v = x @ Wv_eff[b]^T + bv   -> bf16 transposed store [b, d, s]
    launch_gemm(stream, xb, Wve, bv + l * D_, vT, S_, D_, D_, D_, D_, S_,
                1, 0, 0, 0, SD, DD, SD, B_, 1.f, 1 | 2, 4);
    // scores = q @ k^T * 1/8     -> fp32 [b,h,s,s]   (batch = B*H, inner = H)
    launch_gemm(stream, qb, kb, nullptr, scores, S_, S_, HD_, D_, D_, S_,
                H_, HD_, HD_, SS, SD, SD, (long long)H_ * SS, B_ * H_, 0.125f, 0, 4);
    softmax_rows<<<B_ * H_ * S_, 128, 0, stream>>>(scores, attnb);
    // ctx = attn @ vT^T          -> bf16 [b,s,h*64+hd]   (N=64 -> NJ=2 tile)
    launch_gemm(stream, attnb, vT, nullptr, ctxb, S_, HD_, S_, S_, S_, D_,
                H_, SS, (long long)HD_ * S_, HD_, (long long)H_ * SS, (long long)D_ * S_, SD,
                B_ * H_, 1.f, 1, 2);
    // y = ctx @ Wo^T + bo        -> fp32
    launch_gemm(stream, ctxb, Wo16, bo + l * D_, y, S_, D_, D_, D_, D_, D_,
                1, 0, 0, 0, SD, 0, SD, B_, 1.f, 0, 4);
    ln_residual<<<B_ * S_, 256, 0, stream>>>(x, y, g1 + l * D_, be1 + l * D_, xb);
    // h = gelu(x @ W1 + b1)      -> bf16 [b,s,ff]  (W1T is [FF,D])
    launch_gemm(stream, xb, W1T, b1 + l * FF_, hb, S_, FF_, D_, D_, D_, FF_,
                1, 0, 0, 0, SD, 0, (long long)S_ * FF_, B_, 1.f, 1 | 4, 4);
    // y = h @ W2 + b2            -> fp32  (W2T is [D,FF])
    launch_gemm(stream, hb, W2T, b2 + l * D_, y, S_, D_, FF_, FF_, FF_, D_,
                1, 0, 0, 0, (long long)S_ * FF_, 0, SD, B_, 1.f, 0, 4);
    ln_residual<<<B_ * S_, 256, 0, stream>>>(x, y, g2 + l * D_, be2 + l * D_, xb);
  }

  hipMemcpyAsync(d_out, x, (size_t)B_ * S_ * D_ * 4, hipMemcpyDeviceToDevice, stream);
}
